// FactorizedAttention_50714973831489
// MI455X (gfx1250) — matrix operand order repaired and checked by reading, not compiled, not run on hardware
//
#include <hip/hip_runtime.h>
#include <hip/hip_bf16.h>

// ---------------------------------------------------------------------------
// FactorizedAttention on gfx1250: bf16 WMMA GEMMs + VALU softmax/dwconv.
// Round 3: 32x64 wave tiles (8 WMMA / chunk), global_prefetch of k+2 chunk,
// LDS double buffering, branchless clamped edge staging.
// ---------------------------------------------------------------------------

typedef __attribute__((ext_vector_type(16))) __bf16 bf16x16;
typedef __attribute__((ext_vector_type(8)))  float  f32x8;

#define BK 32
#define LDS_PAD 8
#define LDA_S (BK + LDS_PAD)   // 40 bf16 per row -> 80B row stride (bank spread)
#define LDB_S (BK + LDS_PAD)

struct GemmP {
  const float* A;     // M x K row-major (lda)
  const float* B;     // K x N row-major (ldb)
  const float* bias;  // [N] or nullptr
  float*       C;     // M x N row-major (ldc)
  int M, N, K;
  int lda, ldb, ldc;
  long long sAo, sAi; // batch strides: z -> outer = z/inner, in = z%inner
  long long sBo, sBi;
  long long sCo, sCi;
  int inner;
  float alpha;        // scale applied to A during staging
};

template <int BM_, int BN_>
__global__ __launch_bounds__(256)
void gemm_bf16_wmma(GemmP p) {
  constexpr int TT  = BN_ / 32;     // 16-wide col tiles per wave (2 col groups)
  constexpr int MT  = BM_ / 64;     // 16-tall row tiles per wave (4 row groups)
  constexpr int AE  = BM_ / 8;      // A floats staged per thread (8 or 16)
  constexpr int TPR = 32 / AE;      // threads per A row (4 or 2)
  constexpr int GB  = 256 / BN_;    // B k-groups (2 for BN=128, 4 for BN=64)
  constexpr int KC  = BK / GB;      // B k's per thread (16 or 8)

  __shared__ __bf16 As[2][BM_][LDA_S];
  __shared__ __bf16 Bs[2][BN_][LDB_S];  // stored transposed: Bs[n][k]

  const int z     = blockIdx.z;
  const int outer = z / p.inner;
  const int in    = z - outer * p.inner;
  const float* A = p.A + (long long)outer * p.sAo + (long long)in * p.sAi;
  const float* B = p.B + (long long)outer * p.sBo + (long long)in * p.sBi;
  float*       C = p.C + (long long)outer * p.sCo + (long long)in * p.sCi;

  const int m0g = blockIdx.x * BM_;
  const int n0g = blockIdx.y * BN_;

  const int t    = threadIdx.x;
  const int wave = t >> 5;
  const int lane = t & 31;
  const int half = lane >> 4;   // 0: lanes 0-15, 1: lanes 16-31
  const int l16  = lane & 15;

  const int wr = wave & 3;      // row group -> m offset wr*(16*MT)
  const int wc = wave >> 2;     // col group -> n offset wc*(BN_/2)

  // A staging geometry
  const int aRow = t / TPR;
  const int aCol = (t % TPR) * AE;
  // B staging geometry
  const int bN = t % BN_;
  const int bG = t / BN_;

  const bool aFullM = (m0g + BM_ <= p.M);
  const bool bFullN = (n0g + BN_ <= p.N);

  auto stageA = [&](int k0, int buf) {
    if (aFullM && (k0 + BK <= p.K)) {
      const float4* ap = reinterpret_cast<const float4*>(
          A + (long long)(m0g + aRow) * p.lda + k0 + aCol);
      __bf16* d = &As[buf][aRow][aCol];
      #pragma unroll
      for (int v = 0; v < AE / 4; ++v) {
        const float4 f = ap[v];
        d[4 * v + 0] = (__bf16)(f.x * p.alpha);
        d[4 * v + 1] = (__bf16)(f.y * p.alpha);
        d[4 * v + 2] = (__bf16)(f.z * p.alpha);
        d[4 * v + 3] = (__bf16)(f.w * p.alpha);
      }
    } else {
      // clamped addresses, straight-line loads, value-masked (no exec branches)
      int gm = m0g + aRow;
      const bool mOK = gm < p.M;
      if (!mOK) gm = p.M - 1;
      const float* ap = A + (long long)gm * p.lda;
      #pragma unroll
      for (int j = 0; j < AE; ++j) {
        const int gk  = k0 + aCol + j;
        const int gkc = gk < p.K ? gk : p.K - 1;
        float v = ap[gkc] * p.alpha;
        if (!mOK || gk >= p.K) v = 0.0f;
        As[buf][aRow][aCol + j] = (__bf16)v;
      }
    }
  };

  auto stageB = [&](int k0, int buf) {
    if (bFullN && (k0 + BK <= p.K)) {
      const float* bp = B + (long long)(k0 + bG * KC) * p.ldb + n0g + bN;
      #pragma unroll
      for (int j = 0; j < KC; ++j)
        Bs[buf][bN][bG * KC + j] = (__bf16)bp[(long long)j * p.ldb];
    } else {
      int gn = n0g + bN;
      const bool nOK = gn < p.N;
      if (!nOK) gn = p.N - 1;
      #pragma unroll
      for (int j = 0; j < KC; ++j) {
        const int gk  = k0 + bG * KC + j;
        const int gkc = gk < p.K ? gk : p.K - 1;
        float v = B[(long long)gkc * p.ldb + gn];
        if (!nOK || gk >= p.K) v = 0.0f;
        Bs[buf][bN][bG * KC + j] = (__bf16)v;
      }
    }
  };

  f32x8 acc[MT][TT] = {};

  const int nk = (p.K + BK - 1) / BK;
  stageA(0, 0);
  stageB(0, 0);
  __syncthreads();

  for (int it = 0; it < nk; ++it) {
    const int cur = it & 1;
    const int nxt = cur ^ 1;

    if (it + 1 < nk) {            // prefetch next chunk into the other buffer
      stageA((it + 1) * BK, nxt);
      stageB((it + 1) * BK, nxt);
    }
    if (it + 2 < nk) {            // warm L2/L0 for the chunk after that
      const int kp = (it + 2) * BK;
      __builtin_prefetch(A + (long long)(m0g + aRow) * p.lda + kp + aCol, 0, 0);
      __builtin_prefetch(B + (long long)(kp + bG * KC) * p.ldb + n0g + bN, 0, 0);
    }

    // ---- MT A fragments per CDNA5 16-bit A layout (16x32, MxK) ----
    bf16x16 af[MT];
    #pragma unroll
    for (int mm = 0; mm < MT; ++mm) {
      const int m = wr * (16 * MT) + mm * 16 + l16;
      const __bf16* pa0 = &As[cur][m][8 * half];        // K = 8*half + 0..7
      const __bf16* pa1 = &As[cur][m][16 + 8 * half];   // K = 16 + 8*half + ..
      #pragma unroll
      for (int j = 0; j < 8; ++j) af[mm][j]     = pa0[j];
      #pragma unroll
      for (int j = 0; j < 8; ++j) af[mm][8 + j] = pa1[j];
    }

    // ---- TT B fragments, each reused by MT WMMAs ----
    #pragma unroll
    for (int tt = 0; tt < TT; ++tt) {
      bf16x16 bfr;
      const int n = wc * (BN_ / 2) + tt * 16 + l16;
      const __bf16* pb = &Bs[cur][n][8 * half];         // K = 8*half + 0..7, 16 + 8*half + 0..7
      #pragma unroll
      for (int j = 0; j < 8; ++j) bfr[j] = pb[j];
      #pragma unroll
      for (int j = 0; j < 8; ++j) bfr[8 + j] = pb[16 + j];
      #pragma unroll
      for (int mm = 0; mm < MT; ++mm) {
        acc[mm][tt] = __builtin_amdgcn_wmma_f32_16x16x32_bf16(
            /*neg_a=*/false, af[mm], /*neg_b=*/false, bfr,
            /*c_mod=*/(short)0, acc[mm][tt], /*reuse_a=*/false, /*reuse_b=*/false);
      }
    }
    __syncthreads();
  }

  // ---- writeback: C/D layout VGPR r -> M = r + 8*half, N = lane&15 ----
  #pragma unroll
  for (int mm = 0; mm < MT; ++mm) {
    #pragma unroll
    for (int tt = 0; tt < TT; ++tt) {
      const int n = n0g + wc * (BN_ / 2) + tt * 16 + l16;
      if (n >= p.N) continue;
      const float bv = p.bias ? p.bias[n] : 0.0f;
      #pragma unroll
      for (int r = 0; r < 8; ++r) {
        const int m = m0g + wr * (16 * MT) + mm * 16 + r + 8 * half;
        if (m < p.M) C[(long long)m * p.ldc + n] = acc[mm][tt][r] + bv;
      }
    }
  }
}

// ---------------------------------------------------------------------------
// softmax over tokens for K: qkv (b,n,1536), k slice at +512. One block per
// (b,h,d) row of length 785; output ksm row stride 800 (float4-aligned rows).
// ---------------------------------------------------------------------------
#define KSM_LD 800

__global__ __launch_bounds__(256)
void softmax_k(const float* __restrict__ qkv, float* __restrict__ ksm) {
  const int row = blockIdx.x;              // (b*8+h)*64+d, 0..32767
  const int d = row & 63;
  const int h = (row >> 6) & 7;
  const int b = row >> 9;
  const float* src = qkv + (long long)b * 785 * 1536 + 512 + h * 64 + d;
  float* dst = ksm + (long long)row * KSM_LD;

  __shared__ float red[256];
  const int t = threadIdx.x;

  float mx = -3.4e38f;
  for (int n = t; n < 785; n += 256)
    mx = fmaxf(mx, src[(long long)n * 1536]);
  red[t] = mx; __syncthreads();
  for (int s = 128; s > 0; s >>= 1) {
    if (t < s) red[t] = fmaxf(red[t], red[t + s]);
    __syncthreads();
  }
  mx = red[0]; __syncthreads();

  float sum = 0.0f;
  for (int n = t; n < 785; n += 256) {
    const float e = __expf(src[(long long)n * 1536] - mx);
    dst[n] = e;
    sum += e;
  }
  red[t] = sum; __syncthreads();
  for (int s = 128; s > 0; s >>= 1) {
    if (t < s) red[t] += red[t + s];
    __syncthreads();
  }
  const float inv = 1.0f / red[0];
  for (int n = t; n < 785; n += 256)
    dst[n] *= inv;
}

// ---------------------------------------------------------------------------
// Depthwise convs on v (tokens 1..784 viewed as (b, 512, 28, 28)).
// Channels: [0,128)->3x3, [128,320)->5x5, [320,512)->7x7.
// Output y[(b*8+h)*785 + n][e], with n==0 row zeroed.
// ---------------------------------------------------------------------------
__global__ __launch_bounds__(256)
void dwconv_kernel(const float* __restrict__ qkv,
                   const float* __restrict__ w3, const float* __restrict__ b3,
                   const float* __restrict__ w5, const float* __restrict__ b5,
                   const float* __restrict__ w7, const float* __restrict__ b7,
                   float* __restrict__ y) {
  const long long idx = (long long)blockIdx.x * 256 + threadIdx.x;
  const long long total = 64LL * 8 * 785 * 64;
  if (idx >= total) return;

  const int e  = (int)(idx & 63);
  long long tmp = idx >> 6;
  const int n  = (int)(tmp % 785);
  tmp /= 785;
  const int h  = (int)(tmp & 7);
  const int b  = (int)(tmp >> 3);
  const int c  = h * 64 + e;

  float out = 0.0f;
  if (n != 0) {
    const int s  = n - 1;
    const int yy = s / 28;
    const int xx = s - yy * 28;

    const float* w; const float* bb; int ksz; int cl;
    if (c < 128)      { w = w3; bb = b3; ksz = 3; cl = c; }
    else if (c < 320) { w = w5; bb = b5; ksz = 5; cl = c - 128; }
    else              { w = w7; bb = b7; ksz = 7; cl = c - 320; }
    const int pad = ksz >> 1;

    float acc = bb[cl];
    const float* wbase = w + (long long)cl * ksz * ksz;
    const float* vbase = qkv + (long long)b * 785 * 1536 + 1024 + c;
    for (int dy = 0; dy < ksz; ++dy) {
      const int iy = yy + dy - pad;
      if (iy < 0 || iy >= 28) continue;
      for (int dx = 0; dx < ksz; ++dx) {
        const int ix = xx + dx - pad;
        if (ix < 0 || ix >= 28) continue;
        const int nn = 1 + iy * 28 + ix;
        acc += wbase[dy * ksz + dx] * vbase[(long long)nn * 1536];
      }
    }
    out = acc;
  }
  y[((long long)(b * 8 + h) * 785 + n) * 64 + e] = out;
}

// ---------------------------------------------------------------------------
// out1[b,n,c] += q[b,n,c] * y[b,h,n,e]   (rel_pos term), c = h*64+e
// ---------------------------------------------------------------------------
__global__ __launch_bounds__(256)
void reladd_kernel(const float* __restrict__ qkv,
                   const float* __restrict__ y,
                   float* __restrict__ out1) {
  const long long idx = (long long)blockIdx.x * 256 + threadIdx.x;
  const long long total = 64LL * 785 * 512;
  if (idx >= total) return;

  const int c = (int)(idx & 511);
  long long tmp = idx >> 9;
  const int n = (int)(tmp % 785);
  const int b = (int)(tmp / 785);
  const int h = c >> 6;
  const int e = c & 63;

  const float q  = qkv[((long long)b * 785 + n) * 1536 + c];
  const float yv = y[((long long)(b * 8 + h) * 785 + n) * 64 + e];
  out1[idx] += q * yv;
}

// ---------------------------------------------------------------------------
// Launch pipeline
// ---------------------------------------------------------------------------
extern "C" void kernel_launch(void* const* d_in, const int* in_sizes, int n_in,
                              void* d_out, int out_size, void* d_ws, size_t ws_size,
                              hipStream_t stream) {
  const float* x     = (const float*)d_in[0];
  const float* W_qkv = (const float*)d_in[1];
  const float* b_qkv = (const float*)d_in[2];
  const float* w3    = (const float*)d_in[3];
  const float* b3    = (const float*)d_in[4];
  const float* w5    = (const float*)d_in[5];
  const float* b5    = (const float*)d_in[6];
  const float* w7    = (const float*)d_in[7];
  const float* b7    = (const float*)d_in[8];
  const float* W_out = (const float*)d_in[9];
  const float* b_out = (const float*)d_in[10];

  float* ws   = (float*)d_ws;
  float* qkv  = ws;                          // 50240*1536
  float* ksm  = qkv  + 50240LL * 1536;       // 32768*800 (padded rows)
  float* ctx  = ksm  + 32768LL * KSM_LD;     // 512*4096
  float* yb   = ctx  + 512LL * 4096;         // (b,h,n,e) = 64*8*785*64
  float* out1 = yb   + 64LL * 8 * 785 * 64;  // 50240*512
  float* out  = (float*)d_out;

  // 1) qkv = x @ W_qkv + b_qkv   (50240 x 512 x 1536)
  {
    GemmP p;
    p.A = x; p.B = W_qkv; p.bias = b_qkv; p.C = qkv;
    p.M = 50240; p.N = 1536; p.K = 512;
    p.lda = 512; p.ldb = 1536; p.ldc = 1536;
    p.sAo = 0; p.sAi = 0; p.sBo = 0; p.sBi = 0; p.sCo = 0; p.sCi = 0;
    p.inner = 1; p.alpha = 1.0f;
    gemm_bf16_wmma<128, 128><<<dim3(393, 12, 1), dim3(256), 0, stream>>>(p);
  }

  // 2) k softmax over tokens -> ksm (b,h,d,n), row stride 800
  softmax_k<<<dim3(32768), dim3(256), 0, stream>>>(qkv, ksm);

  // 3) ctx[b,h] = ksm[b,h] (64x785) @ v[b,h] (785x64)
  {
    GemmP p;
    p.A = ksm; p.B = qkv + 1024; p.bias = nullptr; p.C = ctx;
    p.M = 64; p.N = 64; p.K = 785;
    p.lda = KSM_LD; p.ldb = 1536; p.ldc = 64;
    p.sAo = 8LL * 64 * KSM_LD; p.sAi = 64LL * KSM_LD;  // ksm: (b*8+h)*64 rows
    p.sBo = 785LL * 1536;      p.sBi = 64;             // v:   b*785*1536 + h*64
    p.sCo = 8LL * 4096;        p.sCi = 4096;           // ctx: (b*8+h)*64*64
    p.inner = 8; p.alpha = 1.0f;
    gemm_bf16_wmma<64, 64><<<dim3(1, 1, 512), dim3(256), 0, stream>>>(p);
  }

  // 4) depthwise convs -> yb (b,h,n,e), n==0 zeroed
  {
    const long long total = 64LL * 8 * 785 * 64;
    const int grid = (int)((total + 255) / 256);
    dwconv_kernel<<<dim3(grid), dim3(256), 0, stream>>>(qkv, w3, b3, w5, b5, w7, b7, yb);
  }

  // 5) out1[b,h] = (q[b,h] * scale) (785x64) @ ctx[b,h] (64x64)
  {
    GemmP p;
    p.A = qkv; p.B = ctx; p.bias = nullptr; p.C = out1;
    p.M = 785; p.N = 64; p.K = 64;
    p.lda = 1536; p.ldb = 64; p.ldc = 512;
    p.sAo = 785LL * 1536; p.sAi = 64;        // q:    b*785*1536 + h*64
    p.sBo = 8LL * 4096;   p.sBi = 4096;      // ctx
    p.sCo = 785LL * 512;  p.sCi = 64;        // out1: b*785*512 + h*64
    p.inner = 8; p.alpha = 0.125f;           // HD^-0.5 = 1/8
    gemm_bf16_wmma<64, 64><<<dim3(13, 1, 512), dim3(256), 0, stream>>>(p);
  }

  // 6) out1 += q * y   (rel_pos)
  {
    const long long total = 64LL * 785 * 512;
    const int grid = (int)((total + 255) / 256);
    reladd_kernel<<<dim3(grid), dim3(256), 0, stream>>>(qkv, yb, out1);
  }

  // 7) out = out1 @ W_out + b_out  (50240 x 512 x 512)
  {
    GemmP p;
    p.A = out1; p.B = W_out; p.bias = b_out; p.C = out;
    p.M = 50240; p.N = 512; p.K = 512;
    p.lda = 512; p.ldb = 512; p.ldc = 512;
    p.sAo = 0; p.sAi = 0; p.sBo = 0; p.sBi = 0; p.sCo = 0; p.sCi = 0;
    p.inner = 1; p.alpha = 1.0f;
    gemm_bf16_wmma<128, 128><<<dim3(393, 4, 1), dim3(256), 0, stream>>>(p);
  }
}
